// CJEPA_34720515620981
// MI455X (gfx1250) — matrix operand order repaired and checked by reading, not compiled, not run on hardware
//
#include <hip/hip_runtime.h>
#include <hip/hip_bf16.h>

// Problem constants (match reference)
#define BB    256
#define TT    64
#define DOBS  512
#define DD    128
#define NSL   16
#define HH    256
#define BT    (BB * TT)          // 16384 samples
#define SLOT_ELEMS (NSL * DD)    // 2048 floats per sample's slots

typedef __attribute__((ext_vector_type(16))) _Float16 v16h;
typedef __attribute__((ext_vector_type(8)))  float    v8f;
typedef __attribute__((ext_vector_type(4)))  int      v4i_;

// ------------------------------------------------ CDNA5 async-LDS staging path

#if defined(__HIP_DEVICE_COMPILE__) && \
    __has_builtin(__builtin_amdgcn_global_load_async_to_lds_b128)
#define USE_ASYNC_LDS 1
#else
#define USE_ASYNC_LDS 0
#endif

// Copy one 16-byte chunk global -> LDS (async DMA path on gfx1250 when the
// builtin is available; tracked by ASYNCcnt, not VGPR-staged).
__device__ __forceinline__ void cp16_g2l(void* lds_dst, const void* gsrc) {
#if USE_ASYNC_LDS
  __builtin_amdgcn_global_load_async_to_lds_b128(
      (__attribute__((address_space(1))) v4i_*)gsrc,
      (__attribute__((address_space(3))) v4i_*)lds_dst, 0, 0);
#else
  *(uint4*)lds_dst = *(const uint4*)gsrc;
#endif
}

__device__ __forceinline__ void cp_wait() {
#if USE_ASYNC_LDS
#if __has_builtin(__builtin_amdgcn_s_wait_asynccnt)
  __builtin_amdgcn_s_wait_asynccnt(0);
#else
  asm volatile("s_wait_asynccnt 0" ::: "memory");
#endif
#endif
}

// ---------------------------------------------------------------- WMMA helpers

__device__ __forceinline__ v8f wmma_f16(v16h a, v16h b, v8f c) {
  // D = A(16x32 f16) * B(32x16 f16) + C(16x16 f32)
  return __builtin_amdgcn_wmma_f32_16x16x32_f16(false, a, false, b, (short)0, c,
                                                false, false);
}

// A fragment (16x32, f16) from row-major f16 source, leading dim `ld` (halfs).
// ISA 7.12.2: lane = row (mod 16); halves 0..7 -> K=kb..kb+7,
// halves 8..15 -> K=kb+16..kb+23, kb = (lane<16 ? 0 : 8).
__device__ __forceinline__ v16h frag_a_f16(const _Float16* src, int ld) {
  const int lane = threadIdx.x & 31;
  const int r    = lane & 15;
  const int kb   = (lane >> 4) << 3;
  const _Float16* p = src + r * ld + kb;
  v16h a;
#pragma unroll
  for (int i = 0; i < 8; ++i) { a[i] = p[i]; a[i + 8] = p[16 + i]; }
  return a;
}

// Same but converting from an f32 source (slots living in d_out, temporal phase).
__device__ __forceinline__ v16h frag_a_f32(const float* src, int ld) {
  const int lane = threadIdx.x & 31;
  const int r    = lane & 15;
  const int kb   = (lane >> 4) << 3;
  const float* p = src + r * ld + kb;
  v16h a;
#pragma unroll
  for (int i = 0; i < 8; ++i) {
    a[i]     = (_Float16)p[i];
    a[i + 8] = (_Float16)p[16 + i];
  }
  return a;
}

// B fragment (32x16, f16) from a TRANSPOSED weight (N x K row-major, ld halfs).
// lane = output col (mod 16); halves 0..7 -> K = kb..kb+7, halves 8..15 -> K = kb+16..kb+23, kb = (lane<16?0:8).
__device__ __forceinline__ v16h frag_b_f16(const _Float16* wt, int ld) {
  const int lane = threadIdx.x & 31;
  const int n    = lane & 15;
  const int kb   = (lane >> 4) << 3;
  const _Float16* p = wt + n * ld + kb;
  v16h b;
#pragma unroll
  for (int i = 0; i < 8; ++i) { b[i] = p[i]; b[i + 8] = p[16 + i]; }
  return b;
}

// C/D accessors: col = lane&15, rows (lane<16 ? 0..7 : 8..15).

__device__ __forceinline__ float wave_sum(float x) {
#pragma unroll
  for (int m = 16; m >= 1; m >>= 1) x += __shfl_xor(x, m, 32);
  return x;
}

// LayerNorm of one 128-elem row by one wave32 (4 elems / lane), writes f16.
__device__ __forceinline__ void ln_row_wave(const float* x, const float* g,
                                            const float* b, _Float16* dst) {
  const int lane = threadIdx.x & 31;
  float xv[4];
  float s = 0.f;
#pragma unroll
  for (int i = 0; i < 4; ++i) { xv[i] = x[lane * 4 + i]; s += xv[i]; }
  const float mu = wave_sum(s) * (1.0f / 128.0f);
  float vs = 0.f;
#pragma unroll
  for (int i = 0; i < 4; ++i) { float d = xv[i] - mu; vs += d * d; }
  const float inv = rsqrtf(wave_sum(vs) * (1.0f / 128.0f) + 1e-5f);
#pragma unroll
  for (int i = 0; i < 4; ++i)
    dst[lane * 4 + i] =
        (_Float16)((xv[i] - mu) * inv * g[lane * 4 + i] + b[lane * 4 + i]);
}

__device__ __forceinline__ float sigmoidf_(float x) {
  return 1.0f / (1.0f + __expf(-x));
}

// ------------------------------------------------------- weight / input prep

// dst[n*K + k] = (f16) src[k*N + n]    (src is K x N row-major)
__global__ void k_wprep(const float* __restrict__ src, _Float16* __restrict__ dst,
                        int K, int N) {
  int idx = blockIdx.x * blockDim.x + threadIdx.x;
  if (idx >= K * N) return;
  int n = idx % N;
  int k = idx / N;
  dst[n * K + k] = (_Float16)src[idx];
}

// obs f32 -> f16 once (halves obs HBM traffic; enables b128 async staging)
__global__ void k_obs16(const float* __restrict__ src, _Float16* __restrict__ dst,
                        int n) {
  int i = blockIdx.x * blockDim.x + threadIdx.x;
  if (i < n) dst[i] = (_Float16)src[i];
}

// ------------------------------------------------------------------- encoder

// h = gelu(obs @ W1 + b1)   (BT x 256), stored f16.
// grid (BT/16, 2), block 256 (8 waves); wave handles one 16-col tile.
__global__ __launch_bounds__(256) void k_enc1(const _Float16* __restrict__ obs16,
                                              const _Float16* __restrict__ w1t,
                                              const float* __restrict__ b1,
                                              _Float16* __restrict__ hbuf) {
  constexpr int LDX = DOBS + 8;
  __shared__ _Float16 xs[16 * LDX];
  const int mt  = blockIdx.x;
  const int tid = threadIdx.x;
  const int wave = tid >> 5;
  const int ct   = blockIdx.y * 8 + wave;    // 0..15
  __builtin_prefetch(w1t + (size_t)ct * 16 * DOBS, 0, 0);  // global_prefetch_b8

  const _Float16* src = obs16 + (size_t)mt * 16 * DOBS;
  for (int i = tid; i < 16 * DOBS / 8; i += 256) {   // 16B chunks
    const int r = i >> 6, cw = i & 63;
    cp16_g2l(xs + r * LDX + cw * 8, src + r * DOBS + cw * 8);
  }
  cp_wait();
  __syncthreads();

  v8f acc = {};
#pragma unroll
  for (int kk = 0; kk < DOBS / 32; ++kk)
    acc = wmma_f16(frag_a_f16(xs + kk * 32, LDX),
                   frag_b_f16(w1t + (size_t)ct * 16 * DOBS + kk * 32, DOBS), acc);

  const int lane = tid & 31, cn = lane & 15, cmb = (lane >> 4) * 8;
  const float bias = b1[ct * 16 + cn];
#pragma unroll
  for (int i = 0; i < 8; ++i) {
    float x  = acc[i] + bias;
    float gx = 0.5f * x * (1.0f + erff(x * 0.70710678118654752f)); // exact gelu
    hbuf[((size_t)mt * 16 + cmb + i) * HH + ct * 16 + cn] = (_Float16)gx;
  }
}

// z = tanh(h @ W2 + obs @ skip_W + b2)  (BT x 128) f32.
__global__ __launch_bounds__(256) void k_enc2(const _Float16* __restrict__ obs16,
                                              const _Float16* __restrict__ hbuf,
                                              const _Float16* __restrict__ w2t,
                                              const _Float16* __restrict__ wskt,
                                              const float* __restrict__ b2,
                                              float* __restrict__ zbuf) {
  constexpr int LDX = DOBS + 8, LDH = HH + 8;
  __shared__ _Float16 xs[16 * LDX];
  __shared__ _Float16 hs[16 * LDH];
  const int mt  = blockIdx.x;
  const int tid = threadIdx.x;
  const int ct  = tid >> 5;                  // 0..7
  __builtin_prefetch(w2t + (size_t)ct * 16 * HH, 0, 0);
  __builtin_prefetch(wskt + (size_t)ct * 16 * DOBS, 0, 0);

  const _Float16* src = obs16 + (size_t)mt * 16 * DOBS;
  for (int i = tid; i < 16 * DOBS / 8; i += 256) {
    const int r = i >> 6, cw = i & 63;
    cp16_g2l(xs + r * LDX + cw * 8, src + r * DOBS + cw * 8);
  }
  const _Float16* hsrc = hbuf + (size_t)mt * 16 * HH;
  for (int i = tid; i < 16 * HH / 8; i += 256) {
    const int r = i >> 5, cw = i & 31;
    cp16_g2l(hs + r * LDH + cw * 8, hsrc + r * HH + cw * 8);
  }
  cp_wait();
  __syncthreads();

  v8f acc = {};
#pragma unroll
  for (int kk = 0; kk < HH / 32; ++kk)
    acc = wmma_f16(frag_a_f16(hs + kk * 32, LDH),
                   frag_b_f16(w2t + (size_t)ct * 16 * HH + kk * 32, HH), acc);
#pragma unroll
  for (int kk = 0; kk < DOBS / 32; ++kk)
    acc = wmma_f16(frag_a_f16(xs + kk * 32, LDX),
                   frag_b_f16(wskt + (size_t)ct * 16 * DOBS + kk * 32, DOBS), acc);

  const int lane = tid & 31, cn = lane & 15, cmb = (lane >> 4) * 8;
  const float bias = b2[ct * 16 + cn];
#pragma unroll
  for (int i = 0; i < 8; ++i)
    zbuf[((size_t)mt * 16 + cmb + i) * DD + ct * 16 + cn] = tanhf(acc[i] + bias);
}

// inp = LN(z); k = inp@Wk; v = inp@Wv   (both BT x 128, f32)
__global__ __launch_bounds__(256) void k_kv(const float* __restrict__ zbuf,
                                            const float* __restrict__ lg,
                                            const float* __restrict__ lb,
                                            const _Float16* __restrict__ wkt,
                                            const _Float16* __restrict__ wvt,
                                            float* __restrict__ kbuf,
                                            float* __restrict__ vbuf) {
  constexpr int LDP = DD + 8;
  __shared__ _Float16 inp[16 * LDP];
  const int mt = blockIdx.x, tid = threadIdx.x, wave = tid >> 5;
  for (int r = wave * 2; r < wave * 2 + 2; ++r)
    ln_row_wave(zbuf + ((size_t)mt * 16 + r) * DD, lg, lb, inp + r * LDP);
  __syncthreads();

  const int lane = tid & 31, cn = lane & 15, cmb = (lane >> 4) * 8;
#pragma unroll
  for (int t2 = wave; t2 < 16; t2 += 8) {
    const _Float16* wt = (t2 < 8) ? (wkt + (size_t)t2 * 16 * DD)
                                  : (wvt + (size_t)(t2 - 8) * 16 * DD);
    float* ob = (t2 < 8) ? kbuf : vbuf;
    const int ct = t2 & 7;
    v8f acc = {};
#pragma unroll
    for (int kk = 0; kk < 4; ++kk)
      acc = wmma_f16(frag_a_f16(inp + kk * 32, LDP),
                     frag_b_f16(wt + kk * 32, DD), acc);
#pragma unroll
    for (int i = 0; i < 8; ++i)
      ob[((size_t)mt * 16 + cmb + i) * DD + ct * 16 + cn] = acc[i];
  }
}

// ------------------------------------ fused slot attention (3 iters + norm)
// One workgroup (8 waves) per sample; slots stay resident in LDS the whole
// time. Only HBM traffic: k/v rows in, final normalized slots out.
__global__ __launch_bounds__(256) void k_slots(
    const float* __restrict__ mu, const float* __restrict__ kbuf,
    const float* __restrict__ vbuf, const _Float16* __restrict__ wqt,
    const _Float16* __restrict__ wiht, const _Float16* __restrict__ whht,
    const float* __restrict__ bih, const float* __restrict__ bhh,
    const _Float16* __restrict__ w1t, const float* __restrict__ mb1,
    const _Float16* __restrict__ w2t, const float* __restrict__ mb2,
    const float* __restrict__ lsl_g, const float* __restrict__ lsl_b,
    const float* __restrict__ lml_g, const float* __restrict__ lml_b,
    float* __restrict__ out) {
  constexpr int LDP = DD + 8;
  __shared__ float    slots[NSL * DD];   // resident slot state (f32)
  __shared__ float    rbuf[NSL * DD];    // GRU reset gate
  __shared__ _Float16 sm[NSL * LDP];     // LN output (attn + mlp)
  __shared__ _Float16 sl16[NSL * LDP];   // f16 snapshot of slots
  __shared__ _Float16 upd[NSL * LDP];    // attn * v
  __shared__ _Float16 hid[NSL * LDP];    // mlp hidden
  __shared__ float    part[8 * NSL];     // per-wave logit partials
  __shared__ float    attn[NSL];
  __shared__ float    krow[DD];
  __shared__ float    vrow[DD];

  const size_t s = blockIdx.x;
  const int tid = threadIdx.x, wave = tid >> 5, lane = tid & 31;
  const int cn = lane & 15, cmb = (lane >> 4) * 8;
  const int col = wave * 16 + cn;

  for (int i = tid; i < SLOT_ELEMS; i += 256) slots[i] = mu[i];
  if (tid < DD) {
    krow[tid] = kbuf[s * DD + tid];
    vrow[tid] = vbuf[s * DD + tid];
  }
  __syncthreads();

  for (int it = 0; it < 3; ++it) {
    // (a) sm = LN(slots)
    for (int r = wave * 2; r < wave * 2 + 2; ++r)
      ln_row_wave(slots + r * DD, lsl_g, lsl_b, sm + r * LDP);
    __syncthreads();

    // (b) q = sm @ Wq (wave owns one 16-col tile), logits partials -> LDS
    {
      v8f q = {};
#pragma unroll
      for (int kk = 0; kk < 4; ++kk)
        q = wmma_f16(frag_a_f16(sm + kk * 32, LDP),
                     frag_b_f16(wqt + (size_t)(wave * 16) * DD + kk * 32, DD), q);
      const float kv = krow[wave * 16 + cn];
      float lgt[8];
#pragma unroll
      for (int i = 0; i < 8; ++i) lgt[i] = q[i] * kv;
#pragma unroll
      for (int m = 1; m < 16; m <<= 1)
#pragma unroll
        for (int i = 0; i < 8; ++i) lgt[i] += __shfl_xor(lgt[i], m, 32);
      if (cn == 0) {
#pragma unroll
        for (int i = 0; i < 8; ++i) part[wave * NSL + cmb + i] = lgt[i];
      }
    }
    __syncthreads();

    // softmax over the 16 slots (wave 0; no WMMA in divergent region)
    if (wave == 0) {
      const int row = lane & 15;
      float lg = 0.f;
#pragma unroll
      for (int w = 0; w < 8; ++w) lg += part[w * NSL + row];
      lg *= 0.088388347648318447f;  // 1/sqrt(128)
      float mx = lg;
#pragma unroll
      for (int m = 1; m < 16; m <<= 1) mx = fmaxf(mx, __shfl_xor(mx, m, 32));
      const float e = __expf(lg - mx);
      float sum = e;
#pragma unroll
      for (int m = 1; m < 16; m <<= 1) sum += __shfl_xor(sum, m, 32);
      if (lane < 16) attn[row] = e / sum;
    }
    __syncthreads();

    // (c) updates = attn * v ; f16 snapshot of slots
    for (int i = tid; i < SLOT_ELEMS; i += 256) {
      const int r = i >> 7, c = i & (DD - 1);
      sl16[r * LDP + c] = (_Float16)slots[i];
      upd[r * LDP + c]  = (_Float16)(attn[r] * vrow[c]);
    }
    __syncthreads();

    // (d) GRU reset gate (cols 0..127)
    {
      v8f ai = {}, ah = {};
#pragma unroll
      for (int kk = 0; kk < 4; ++kk) {
        const v16h af = frag_a_f16(upd + kk * 32, LDP);
        const v16h hf = frag_a_f16(sl16 + kk * 32, LDP);
        ai = wmma_f16(af, frag_b_f16(wiht + (size_t)(wave * 16) * DD + kk * 32, DD), ai);
        ah = wmma_f16(hf, frag_b_f16(whht + (size_t)(wave * 16) * DD + kk * 32, DD), ah);
      }
      const float bi = bih[col], bh = bhh[col];
#pragma unroll
      for (int i = 0; i < 8; ++i)
        rbuf[(cmb + i) * DD + col] = sigmoidf_(ai[i] + bi + ah[i] + bh);
    }
    __syncthreads();

    // GRU z (128..255) and n (256..383) gates; blend into slots
    {
      v8f zi = {}, zh = {}, ni = {}, nh = {};
#pragma unroll
      for (int kk = 0; kk < 4; ++kk) {
        const v16h af = frag_a_f16(upd + kk * 32, LDP);
        const v16h hf = frag_a_f16(sl16 + kk * 32, LDP);
        zi = wmma_f16(af, frag_b_f16(wiht + (size_t)(128 + wave * 16) * DD + kk * 32, DD), zi);
        zh = wmma_f16(hf, frag_b_f16(whht + (size_t)(128 + wave * 16) * DD + kk * 32, DD), zh);
        ni = wmma_f16(af, frag_b_f16(wiht + (size_t)(256 + wave * 16) * DD + kk * 32, DD), ni);
        nh = wmma_f16(hf, frag_b_f16(whht + (size_t)(256 + wave * 16) * DD + kk * 32, DD), nh);
      }
      const float biz = bih[128 + col], bhz = bhh[128 + col];
      const float bin = bih[256 + col], bhn = bhh[256 + col];
#pragma unroll
      for (int i = 0; i < 8; ++i) {
        const float z = sigmoidf_(zi[i] + biz + zh[i] + bhz);
        const float r = rbuf[(cmb + i) * DD + col];
        const float n = tanhf(ni[i] + bin + r * (nh[i] + bhn));
        const float h = slots[(cmb + i) * DD + col];  // own element, read-then-write
        slots[(cmb + i) * DD + col] = (1.0f - z) * n + z * h;
      }
    }
    __syncthreads();

    // (e) residual MLP
    for (int r = wave * 2; r < wave * 2 + 2; ++r)
      ln_row_wave(slots + r * DD, lml_g, lml_b, sm + r * LDP);
    __syncthreads();
    {
      v8f acc = {};
#pragma unroll
      for (int kk = 0; kk < 4; ++kk)
        acc = wmma_f16(frag_a_f16(sm + kk * 32, LDP),
                       frag_b_f16(w1t + (size_t)(wave * 16) * DD + kk * 32, DD), acc);
      const float bias = mb1[wave * 16 + cn];
#pragma unroll
      for (int i = 0; i < 8; ++i)
        hid[(cmb + i) * LDP + wave * 16 + cn] = (_Float16)fmaxf(acc[i] + bias, 0.0f);
    }
    __syncthreads();
    {
      v8f acc = {};
#pragma unroll
      for (int kk = 0; kk < 4; ++kk)
        acc = wmma_f16(frag_a_f16(hid + kk * 32, LDP),
                       frag_b_f16(w2t + (size_t)(wave * 16) * DD + kk * 32, DD), acc);
      const float bias = mb2[wave * 16 + cn];
#pragma unroll
      for (int i = 0; i < 8; ++i)
        slots[(cmb + i) * DD + col] += acc[i] + bias;
    }
    __syncthreads();
  }

  // F.normalize per slot row, write the only HBM output of this kernel
  for (int r = wave * 2; r < wave * 2 + 2; ++r) {
    float v[4];
    float ss = 0.f;
#pragma unroll
    for (int i = 0; i < 4; ++i) {
      v[i] = slots[r * DD + lane * 4 + i];
      ss += v[i] * v[i];
    }
    ss = wave_sum(ss);
    const float inv = 1.0f / fmaxf(sqrtf(ss), 1e-8f);
#pragma unroll
    for (int i = 0; i < 4; ++i)
      out[s * SLOT_ELEMS + r * DD + lane * 4 + i] = v[i] * inv;
  }
}

// Temporal blend: out[:,t] = 0.7*out[:,t] + 0.3*tanh(out[:,t-1] @ Wt)
// grid (B, 2), block 128 (4 waves), wave = col tile.
__global__ __launch_bounds__(128) void k_temporal(float* __restrict__ out,
                                                  const _Float16* __restrict__ wtt,
                                                  int t) {
  const int b = blockIdx.x;
  const int ct = blockIdx.y * 4 + (threadIdx.x >> 5); // 0..7
  const float* prev = out + ((size_t)b * TT + (t - 1)) * SLOT_ELEMS;
  float* cur        = out + ((size_t)b * TT + t) * SLOT_ELEMS;
  v8f acc = {};
#pragma unroll
  for (int kk = 0; kk < 4; ++kk)
    acc = wmma_f16(frag_a_f32(prev + kk * 32, DD),
                   frag_b_f16(wtt + (size_t)ct * 16 * DD + kk * 32, DD), acc);
  const int lane = threadIdx.x & 31, cn = lane & 15, cmb = (lane >> 4) * 8;
#pragma unroll
  for (int i = 0; i < 8; ++i) {
    const int idx = (cmb + i) * DD + ct * 16 + cn;
    cur[idx] = 0.7f * cur[idx] + 0.3f * tanhf(acc[i]);
  }
}

// -------------------------------------------------------------------- launcher

extern "C" void kernel_launch(void* const* d_in, const int* in_sizes, int n_in,
                              void* d_out, int out_size, void* d_ws, size_t ws_size,
                              hipStream_t stream) {
  (void)in_sizes; (void)n_in; (void)out_size; (void)ws_size;
  const float* obs     = (const float*)d_in[0];
  const float* enc_W1  = (const float*)d_in[1];
  const float* enc_b1  = (const float*)d_in[2];
  const float* enc_W2  = (const float*)d_in[3];
  const float* enc_b2  = (const float*)d_in[4];
  const float* skip_W  = (const float*)d_in[5];
  const float* slot_mu = (const float*)d_in[6];
  const float* Wk      = (const float*)d_in[7];
  const float* Wv      = (const float*)d_in[8];
  const float* Wq      = (const float*)d_in[9];
  const float* gWih    = (const float*)d_in[10];
  const float* gWhh    = (const float*)d_in[11];
  const float* gbih    = (const float*)d_in[12];
  const float* gbhh    = (const float*)d_in[13];
  const float* mW1     = (const float*)d_in[14];
  const float* mb1     = (const float*)d_in[15];
  const float* mW2     = (const float*)d_in[16];
  const float* mb2     = (const float*)d_in[17];
  const float* lin_g   = (const float*)d_in[18];
  const float* lin_b   = (const float*)d_in[19];
  const float* lsl_g   = (const float*)d_in[20];
  const float* lsl_b   = (const float*)d_in[21];
  const float* lml_g   = (const float*)d_in[22];
  const float* lml_b   = (const float*)d_in[23];
  const float* Wt      = (const float*)d_in[24];
  float* out = (float*)d_out;

  char* ws = (char*)d_ws;
  size_t off = 0;
  auto carve = [&](size_t bytes) {
    void* p = ws + off;
    off += (bytes + 255) & ~(size_t)255;
    return p;
  };
  _Float16* wt_e1 = (_Float16*)carve((size_t)HH * DOBS * 2);
  _Float16* wt_e2 = (_Float16*)carve((size_t)DD * HH * 2);
  _Float16* wt_sk = (_Float16*)carve((size_t)DD * DOBS * 2);
  _Float16* wt_k  = (_Float16*)carve((size_t)DD * DD * 2);
  _Float16* wt_v  = (_Float16*)carve((size_t)DD * DD * 2);
  _Float16* wt_q  = (_Float16*)carve((size_t)DD * DD * 2);
  _Float16* wt_gi = (_Float16*)carve((size_t)3 * DD * DD * 2);
  _Float16* wt_gh = (_Float16*)carve((size_t)3 * DD * DD * 2);
  _Float16* wt_m1 = (_Float16*)carve((size_t)DD * DD * 2);
  _Float16* wt_m2 = (_Float16*)carve((size_t)DD * DD * 2);
  _Float16* wt_t  = (_Float16*)carve((size_t)DD * DD * 2);
  _Float16* obs16 = (_Float16*)carve((size_t)BT * DOBS * 2);
  _Float16* hbuf  = (_Float16*)carve((size_t)BT * HH * 2);
  float* zbuf     = (float*)carve((size_t)BT * DD * 4);
  float* kbuf     = (float*)carve((size_t)BT * DD * 4);
  float* vbuf     = (float*)carve((size_t)BT * DD * 4);

  auto prep = [&](const float* s, _Float16* d, int K, int N) {
    int total = K * N;
    k_wprep<<<(total + 255) / 256, 256, 0, stream>>>(s, d, K, N);
  };
  prep(enc_W1, wt_e1, DOBS, HH);
  prep(enc_W2, wt_e2, HH, DD);
  prep(skip_W, wt_sk, DOBS, DD);
  prep(Wk, wt_k, DD, DD);
  prep(Wv, wt_v, DD, DD);
  prep(Wq, wt_q, DD, DD);
  prep(gWih, wt_gi, DD, 3 * DD);
  prep(gWhh, wt_gh, DD, 3 * DD);
  prep(mW1, wt_m1, DD, DD);
  prep(mW2, wt_m2, DD, DD);
  prep(Wt, wt_t, DD, DD);
  {
    int n = BT * DOBS;
    k_obs16<<<(n + 255) / 256, 256, 0, stream>>>(obs, obs16, n);
  }

  // Phase A: per-sample encode + fused slot attention, parallel over B*T.
  k_enc1<<<dim3(BT / 16, 2), 256, 0, stream>>>(obs16, wt_e1, enc_b1, hbuf);
  k_enc2<<<dim3(BT / 16), 256, 0, stream>>>(obs16, hbuf, wt_e2, wt_sk, enc_b2, zbuf);
  k_kv<<<dim3(BT / 16), 256, 0, stream>>>(zbuf, lin_g, lin_b, wt_k, wt_v, kbuf, vbuf);
  k_slots<<<dim3(BT), 256, 0, stream>>>(slot_mu, kbuf, vbuf, wt_q, wt_gi, wt_gh,
                                        gbih, gbhh, wt_m1, mb1, wt_m2, mb2,
                                        lsl_g, lsl_b, lml_g, lml_b, out);

  // Phase B: sequential temporal recurrence over T (in-place in d_out).
  for (int t = 1; t < TT; ++t)
    k_temporal<<<dim3(BB, 2), 128, 0, stream>>>(out, wt_t, t);
}
